// Pointnet2Backbone_50139448213744
// MI455X (gfx1250) — matrix operand order repaired and checked by reading, not compiled, not run on hardware
//
#include <hip/hip_runtime.h>

typedef _Float16 h8  __attribute__((ext_vector_type(8)));
typedef _Float16 h16 __attribute__((ext_vector_type(16)));
typedef float    f8  __attribute__((ext_vector_type(8)));

// ---------------------------------------------------------------------------
// Farthest point sampling: one block per batch. xyz + running min-dists in LDS.
// Matches jax scan: idx[0]=0, then 'npoint-1' iterations of min-update + argmax
// (argmax tie-break = smallest index, like jnp.argmax).
// ---------------------------------------------------------------------------
__global__ __launch_bounds__(256)
void fps_kernel(const float* __restrict__ xyz, int N, int npoint,
                int* __restrict__ fps_idx, float* __restrict__ new_xyz)
{
    __shared__ float sx[4096], sy[4096], sz[4096], sd[4096];
    __shared__ float rv[256];
    __shared__ int   ri[256];
    __shared__ int   sidx[512];
    __shared__ int   s_last;
    const int b   = blockIdx.x;
    const int tid = threadIdx.x;
    const float* base = xyz + (size_t)b * N * 3;
    for (int j = tid; j < N; j += 256) {
        sx[j] = base[j * 3 + 0];
        sy[j] = base[j * 3 + 1];
        sz[j] = base[j * 3 + 2];
        sd[j] = 1e10f;
    }
    if (tid == 0) { sidx[0] = 0; s_last = 0; }
    __syncthreads();
    for (int i = 1; i < npoint; i++) {
        const int last = s_last;
        const float px = sx[last], py = sy[last], pz = sz[last];
        float bestv = -1.0f; int besti = 0;
        for (int j = tid; j < N; j += 256) {
            float dx = sx[j] - px, dy = sy[j] - py, dz = sz[j] - pz;
            float d = fminf(sd[j], dx * dx + dy * dy + dz * dz);
            sd[j] = d;
            if (d > bestv) { bestv = d; besti = j; }   // ascending j keeps earliest on ties
        }
        rv[tid] = bestv; ri[tid] = besti;
        __syncthreads();
        for (int off = 128; off > 0; off >>= 1) {
            if (tid < off) {
                float ov = rv[tid + off]; int oi = ri[tid + off];
                if (ov > rv[tid] || (ov == rv[tid] && oi < ri[tid])) { rv[tid] = ov; ri[tid] = oi; }
            }
            __syncthreads();
        }
        if (tid == 0) { s_last = ri[0]; sidx[i] = ri[0]; }
        __syncthreads();
    }
    for (int m = tid; m < npoint; m += 256) {
        int id = sidx[m];
        fps_idx[(size_t)b * npoint + m] = id;
        float* o = new_xyz + ((size_t)b * npoint + m) * 3;
        o[0] = sx[id]; o[1] = sy[id]; o[2] = sz[id];
    }
}

// ---------------------------------------------------------------------------
// Ball query: serial ascending scan == stable argsort-of-mask semantics.
// ---------------------------------------------------------------------------
__global__ __launch_bounds__(256)
void ballquery_kernel(const float* __restrict__ xyz, int N,
                      const float* __restrict__ new_xyz, int M,
                      int nsample, float r2, int* __restrict__ idx)
{
    __shared__ float sx[4096], sy[4096], sz[4096];
    const int b = blockIdx.y;
    const float* base = xyz + (size_t)b * N * 3;
    for (int j = threadIdx.x; j < N; j += 256) {
        sx[j] = base[j * 3 + 0]; sy[j] = base[j * 3 + 1]; sz[j] = base[j * 3 + 2];
    }
    __syncthreads();
    const int m = blockIdx.x * 256 + threadIdx.x;
    if (m >= M) return;
    const float cx = new_xyz[((size_t)b * M + m) * 3 + 0];
    const float cy = new_xyz[((size_t)b * M + m) * 3 + 1];
    const float cz = new_xyz[((size_t)b * M + m) * 3 + 2];
    int* o = idx + ((size_t)b * M + m) * nsample;
    int cnt = 0, first = 0;
    for (int j = 0; j < N && cnt < nsample; j++) {
        float dx = sx[j] - cx, dy = sy[j] - cy, dz = sz[j] - cz;
        if (dx * dx + dy * dy + dz * dz < r2) {
            if (cnt == 0) first = j;
            o[cnt++] = j;
        }
    }
    for (; cnt < nsample; cnt++) o[cnt] = first;
}

// ---------------------------------------------------------------------------
// Grouping kernels -> zero-padded f16 X buffers (row stride = Kpad).
// ---------------------------------------------------------------------------
__global__ __launch_bounds__(256)
void group_sa1_kernel(const float* __restrict__ xyz, const float* __restrict__ nxz,
                      const int* __restrict__ idx, _Float16* __restrict__ X)
{
    int row = blockIdx.x * 256 + threadIdx.x;          // B*512*32 rows
    if (row >= 32 * 512 * 32) return;
    int bm = row >> 5;                                  // b*512+m
    int b  = bm >> 9;
    int id = idx[row];
    const float* p = xyz + ((size_t)b * 4096 + id) * 3;
    const float* c = nxz + (size_t)bm * 3;
    _Float16* o = X + (size_t)row * 32;
    o[0] = (_Float16)(p[0] - c[0]);
    o[1] = (_Float16)(p[1] - c[1]);
    o[2] = (_Float16)(p[2] - c[2]);
    for (int k = 3; k < 32; k++) o[k] = (_Float16)0.f;
}

__global__ __launch_bounds__(256)
void group_sa2_kernel(const float* __restrict__ nxz1, const _Float16* __restrict__ feats1,
                      const float* __restrict__ nxz2, const int* __restrict__ idx,
                      _Float16* __restrict__ X)
{
    int row = blockIdx.x * 256 + threadIdx.x;          // B*256*16 rows
    if (row >= 32 * 256 * 16) return;
    int bm = row >> 4;                                  // b*256+m
    int b  = bm >> 8;
    int id = idx[row];
    const float* p = nxz1 + ((size_t)b * 512 + id) * 3;
    const float* c = nxz2 + (size_t)bm * 3;
    _Float16* o = X + (size_t)row * 160;                // Kpad=160 (Cin=131)
    o[0] = (_Float16)(p[0] - c[0]);
    o[1] = (_Float16)(p[1] - c[1]);
    o[2] = (_Float16)(p[2] - c[2]);
    const _Float16* f = feats1 + ((size_t)b * 512 + id) * 128;
    for (int cc = 0; cc < 128; cc++) o[3 + cc] = f[cc];
    for (int k = 131; k < 160; k++) o[k] = (_Float16)0.f;
}

__global__ __launch_bounds__(256)
void group_sa3_kernel(const float* __restrict__ nxz2, const _Float16* __restrict__ feats2,
                      _Float16* __restrict__ X)
{
    int row = blockIdx.x * 256 + threadIdx.x;          // B*256 rows
    if (row >= 32 * 256) return;
    const float* p = nxz2 + (size_t)row * 3;
    _Float16* o = X + (size_t)row * 288;                // Kpad=288 (Cin=259)
    o[0] = (_Float16)p[0]; o[1] = (_Float16)p[1]; o[2] = (_Float16)p[2];
    const _Float16* f = feats2 + (size_t)row * 256;
    for (int cc = 0; cc < 256; cc++) o[3 + cc] = f[cc];
    for (int k = 259; k < 288; k++) o[k] = (_Float16)0.f;
}

// ---------------------------------------------------------------------------
// Weight prep: Wt[n][k] = f16(W[k][n]), zero-padded along K to Kpad.
// ---------------------------------------------------------------------------
__global__ __launch_bounds__(256)
void prep_w_kernel(const float* __restrict__ W, _Float16* __restrict__ Wt,
                   int Cin, int Cout, int Kpad)
{
    int i = blockIdx.x * 256 + threadIdx.x;
    if (i >= Cout * Kpad) return;
    int n = i / Kpad, k = i - n * Kpad;
    Wt[(size_t)n * Kpad + k] = (k < Cin) ? (_Float16)W[(size_t)k * Cout + n] : (_Float16)0.f;
}

// ---------------------------------------------------------------------------
// WMMA GEMM: Y[R x Cout] = f16(X[R x Kpad]) @ W + bias, f32 accumulate.
// Wave computes a 16(M) x 64(N) tile: 4x v_wmma_f32_16x16x32_f16 per K-step.
// A as loaded here (16x32 f16, contiguous k, same order as B): lanes0-15 K=0..15, lanes16-31 K=16..31.
// B as loaded here (32x16 f16, contiguous k, same order as A): lanes0-15 K=0..15, lanes16-31 K=16..31; N = lane&15.
// C/D layout: VGPR r -> row r (lanes0-15) / r+8 (lanes16-31), col = lane&15.
// ---------------------------------------------------------------------------
__global__ __launch_bounds__(256)
void gemm_bias_kernel(const _Float16* __restrict__ X, const _Float16* __restrict__ Wt,
                      const float* __restrict__ bias, float* __restrict__ Y,
                      int R, int Kpad, int Cout)
{
    const int lane   = threadIdx.x & 31;
    const int wave   = blockIdx.x * 8 + (threadIdx.x >> 5);
    const int tilesN = Cout >> 6;
    const int mt = wave / tilesN;
    const int nt = wave - mt * tilesN;
    const int m0 = mt << 4;
    const int n0 = nt << 6;
    const int laneHi = lane >> 4;
    const int laneLo = lane & 15;

    const _Float16* aPtr  = X  + (size_t)(m0 + laneLo) * Kpad + laneHi * 16;
    const _Float16* bPtr0 = Wt + (size_t)(n0 +  0 + laneLo) * Kpad + laneHi * 16;
    const _Float16* bPtr1 = Wt + (size_t)(n0 + 16 + laneLo) * Kpad + laneHi * 16;
    const _Float16* bPtr2 = Wt + (size_t)(n0 + 32 + laneLo) * Kpad + laneHi * 16;
    const _Float16* bPtr3 = Wt + (size_t)(n0 + 48 + laneLo) * Kpad + laneHi * 16;

    f8 acc0 = {}, acc1 = {}, acc2 = {}, acc3 = {};
    for (int k0 = 0; k0 < Kpad; k0 += 32) {
        h16 A, B0, B1, B2, B3;
        *(h8*)&A          = *(const h8*)(aPtr  + k0);
        *(((h8*)&A)  + 1) = *(const h8*)(aPtr  + k0 + 8);
        *(h8*)&B0         = *(const h8*)(bPtr0 + k0);
        *(((h8*)&B0) + 1) = *(const h8*)(bPtr0 + k0 + 8);
        *(h8*)&B1         = *(const h8*)(bPtr1 + k0);
        *(((h8*)&B1) + 1) = *(const h8*)(bPtr1 + k0 + 8);
        *(h8*)&B2         = *(const h8*)(bPtr2 + k0);
        *(((h8*)&B2) + 1) = *(const h8*)(bPtr2 + k0 + 8);
        *(h8*)&B3         = *(const h8*)(bPtr3 + k0);
        *(((h8*)&B3) + 1) = *(const h8*)(bPtr3 + k0 + 8);
        acc0 = __builtin_amdgcn_wmma_f32_16x16x32_f16(false, A, false, B0, (short)0, acc0, false, false);
        acc1 = __builtin_amdgcn_wmma_f32_16x16x32_f16(false, A, false, B1, (short)0, acc1, false, false);
        acc2 = __builtin_amdgcn_wmma_f32_16x16x32_f16(false, A, false, B2, (short)0, acc2, false, false);
        acc3 = __builtin_amdgcn_wmma_f32_16x16x32_f16(false, A, false, B3, (short)0, acc3, false, false);
    }
    const int colBase = n0 + laneLo;
    const float b0 = bias[colBase], b1 = bias[colBase + 16];
    const float b2 = bias[colBase + 32], b3 = bias[colBase + 48];
#pragma unroll
    for (int r = 0; r < 8; r++) {
        int row = m0 + r + laneHi * 8;
        float* y = Y + (size_t)row * Cout + colBase;
        y[0]  = acc0[r] + b0;
        y[16] = acc1[r] + b1;
        y[32] = acc2[r] + b2;
        y[48] = acc3[r] + b3;
    }
}

// ---------------------------------------------------------------------------
// BN stats: deterministic two-stage per-channel sum / sumsq reduction.
// ---------------------------------------------------------------------------
__global__ __launch_bounds__(256)
void stats_partial_kernel(const float* __restrict__ Y, int R, int C,
                          float* __restrict__ psum, float* __restrict__ psumsq, int S)
{
    __shared__ float ss[256], sq[256];
    const int cl = threadIdx.x & 63;
    const int rg = threadIdx.x >> 6;
    const int c  = blockIdx.y * 64 + cl;
    const int rowsPer = R / S;
    const int r0 = blockIdx.x * rowsPer;
    float s = 0.f, q = 0.f;
    for (int r = r0 + rg; r < r0 + rowsPer; r += 4) {
        float v = Y[(size_t)r * C + c];
        s += v; q += v * v;
    }
    ss[threadIdx.x] = s; sq[threadIdx.x] = q;
    __syncthreads();
    if (rg == 0) {
        s = ss[cl] + ss[64 + cl] + ss[128 + cl] + ss[192 + cl];
        q = sq[cl] + sq[64 + cl] + sq[128 + cl] + sq[192 + cl];
        psum  [(size_t)blockIdx.x * C + c] = s;
        psumsq[(size_t)blockIdx.x * C + c] = q;
    }
}

__global__ __launch_bounds__(256)
void stats_final_kernel(const float* __restrict__ psum, const float* __restrict__ psumsq,
                        const float* __restrict__ gamma, const float* __restrict__ beta,
                        float* __restrict__ scsh, int C, int R, int S)
{
    int c = blockIdx.x * 256 + threadIdx.x;
    if (c >= C) return;
    float s = 0.f, q = 0.f;
    for (int i = 0; i < S; i++) { s += psum[(size_t)i * C + c]; q += psumsq[(size_t)i * C + c]; }
    float inv  = 1.0f / (float)R;
    float mean = s * inv;
    float var  = q * inv - mean * mean;
    float sc   = gamma[c] * rsqrtf(var + 1e-5f);
    scsh[c]     = sc;
    scsh[C + c] = beta[c] - mean * sc;
}

// Fused BN(affine) + ReLU, f32 -> f16 (C is a power of two).
__global__ __launch_bounds__(256)
void bnrelu_kernel(const float* __restrict__ Y, const float* __restrict__ scsh,
                   _Float16* __restrict__ H, int C, size_t total)
{
    size_t i = (size_t)blockIdx.x * 256 + threadIdx.x;
    if (i >= total) return;
    int c = (int)(i & (size_t)(C - 1));
    float v = Y[i] * scsh[c] + scsh[C + c];
    H[i] = (_Float16)fmaxf(v, 0.f);
}

// Max over the sample axis. C = 1<<cshift.
__global__ __launch_bounds__(256)
void maxpool_h_kernel(const _Float16* __restrict__ H, _Float16* __restrict__ out,
                      int S, int cshift, size_t total)
{
    size_t i = (size_t)blockIdx.x * 256 + threadIdx.x;
    if (i >= total) return;
    int C = 1 << cshift;
    int c = (int)(i & (size_t)(C - 1));
    size_t rowo = i >> cshift;
    const _Float16* p = H + ((size_t)rowo * S) * C + c;
    float m = -1e30f;
    for (int s = 0; s < S; s++) m = fmaxf(m, (float)p[(size_t)s * C]);
    out[i] = (_Float16)m;
}

__global__ __launch_bounds__(256)
void maxpool_f_kernel(const _Float16* __restrict__ H, float* __restrict__ out,
                      int S, int cshift, size_t total)
{
    size_t i = (size_t)blockIdx.x * 256 + threadIdx.x;
    if (i >= total) return;
    int C = 1 << cshift;
    int c = (int)(i & (size_t)(C - 1));
    size_t rowo = i >> cshift;
    const _Float16* p = H + ((size_t)rowo * S) * C + c;
    float m = -1e30f;
    for (int s = 0; s < S; s++) m = fmaxf(m, (float)p[(size_t)s * C]);
    out[i] = m;
}

// ---------------------------------------------------------------------------
extern "C" void kernel_launch(void* const* d_in, const int* in_sizes, int n_in,
                              void* d_out, int out_size, void* d_ws, size_t ws_size,
                              hipStream_t stream)
{
    (void)in_sizes; (void)n_in; (void)out_size;
    const float* pc = (const float*)d_in[0];
    const float *W[9], *Bb[9], *G[9], *Be[9];
    for (int l = 0; l < 9; l++) {
        W[l]  = (const float*)d_in[1 + l * 4 + 0];
        Bb[l] = (const float*)d_in[1 + l * 4 + 1];
        G[l]  = (const float*)d_in[1 + l * 4 + 2];
        Be[l] = (const float*)d_in[1 + l * 4 + 3];
    }
    static const int CinA[9]  = {3, 64, 64, 131, 128, 128, 259, 256, 512};
    static const int CoutA[9] = {64, 64, 128, 128, 128, 256, 256, 512, 1024};
    static const int KpadA[9] = {32, 64, 64, 160, 128, 128, 288, 256, 512};

    // ---- workspace layout ----
    char* base = (char*)d_ws;
    size_t off = 0;
    auto take = [&](size_t bytes) -> char* {
        char* r = base + off;
        off = (off + bytes + 255) & ~(size_t)255;
        return r;
    };
    _Float16* HA = (_Float16*)take((size_t)524288 * 128 * 2);   // 128 MiB ping
    _Float16* HB = (_Float16*)take((size_t)524288 * 128 * 2);   // 128 MiB pong
    float*    Yb = (float*)   take((size_t)524288 * 128 * 4);   // 256 MiB GEMM out
    _Float16* Wt[9];
    for (int l = 0; l < 9; l++) Wt[l] = (_Float16*)take((size_t)CoutA[l] * KpadA[l] * 2);
    float* psum   = (float*)take((size_t)128 * 1024 * 4);
    float* psumsq = (float*)take((size_t)128 * 1024 * 4);
    float* scsh   = (float*)take((size_t)2048 * 4);
    int*   fps1 = (int*)  take((size_t)32 * 512 * 4);
    float* nxz1 = (float*)take((size_t)32 * 512 * 3 * 4);
    int*   idx1 = (int*)  take((size_t)32 * 512 * 32 * 4);
    int*   fps2 = (int*)  take((size_t)32 * 256 * 4);
    float* nxz2 = (float*)take((size_t)32 * 256 * 3 * 4);
    int*   idx2 = (int*)  take((size_t)32 * 256 * 16 * 4);
    _Float16* feats1 = (_Float16*)take((size_t)32 * 512 * 128 * 2);
    _Float16* feats2 = (_Float16*)take((size_t)32 * 256 * 256 * 2);
    if (off > ws_size) return;  // workspace too small; bail safely

    // ---- weight prep ----
    for (int l = 0; l < 9; l++) {
        int tot = CoutA[l] * KpadA[l];
        prep_w_kernel<<<(tot + 255) / 256, 256, 0, stream>>>(W[l], Wt[l], CinA[l], CoutA[l], KpadA[l]);
    }

    auto run_layer = [&](int l, const _Float16* Xp, _Float16* Hout, int R) {
        const int C = CoutA[l], Kp = KpadA[l];
        const int waves = (R / 16) * (C / 64);
        gemm_bias_kernel<<<waves / 8, 256, 0, stream>>>(Xp, Wt[l], Bb[l], Yb, R, Kp, C);
        dim3 gs(128, C / 64);
        stats_partial_kernel<<<gs, 256, 0, stream>>>(Yb, R, C, psum, psumsq, 128);
        stats_final_kernel<<<(C + 255) / 256, 256, 0, stream>>>(psum, psumsq, G[l], Be[l], scsh, C, R, 128);
        size_t total = (size_t)R * C;
        bnrelu_kernel<<<(unsigned)(total / 256), 256, 0, stream>>>(Yb, scsh, Hout, C, total);
    };

    // ---- SA1: npoint=512, r=0.04, nsample=32, MLP 3->64->64->128 ----
    fps_kernel<<<32, 256, 0, stream>>>(pc, 4096, 512, fps1, nxz1);
    ballquery_kernel<<<dim3(2, 32), 256, 0, stream>>>(pc, 4096, nxz1, 512, 32, 0.04f * 0.04f, idx1);
    group_sa1_kernel<<<524288 / 256, 256, 0, stream>>>(pc, nxz1, idx1, HA);
    run_layer(0, HA, HB, 524288);
    run_layer(1, HB, HA, 524288);
    run_layer(2, HA, HB, 524288);
    maxpool_h_kernel<<<(32 * 512 * 128) / 256, 256, 0, stream>>>(HB, feats1, 32, 7, (size_t)32 * 512 * 128);

    // ---- SA2: npoint=256, r=0.1, nsample=16, MLP 131->128->128->256 ----
    fps_kernel<<<32, 256, 0, stream>>>(nxz1, 512, 256, fps2, nxz2);
    ballquery_kernel<<<dim3(1, 32), 256, 0, stream>>>(nxz1, 512, nxz2, 256, 16, 0.1f * 0.1f, idx2);
    group_sa2_kernel<<<131072 / 256, 256, 0, stream>>>(nxz1, feats1, nxz2, idx2, HA);
    run_layer(3, HA, HB, 131072);
    run_layer(4, HB, HA, 131072);
    run_layer(5, HA, HB, 131072);
    maxpool_h_kernel<<<(32 * 256 * 256) / 256, 256, 0, stream>>>(HB, feats2, 16, 8, (size_t)32 * 256 * 256);

    // ---- SA3: global, MLP 259->256->512->1024, max over 256 -> d_out ----
    group_sa3_kernel<<<8192 / 256, 256, 0, stream>>>(nxz2, feats2, HA);
    run_layer(6, HA, HB, 8192);
    run_layer(7, HB, HA, 8192);
    run_layer(8, HA, HB, 8192);
    maxpool_f_kernel<<<(32 * 1024) / 256, 256, 0, stream>>>(HB, (float*)d_out, 256, 10, (size_t)32 * 1024);
}
